// Attention_33423435497672
// MI455X (gfx1250) — matrix operand order repaired and checked by reading, not compiled, not run on hardware
//
#include <hip/hip_runtime.h>
#include <hip/hip_bf16.h>

typedef __attribute__((ext_vector_type(16))) _Float16 v16h;
typedef __attribute__((ext_vector_type(8)))  _Float16 v8h;
typedef __attribute__((ext_vector_type(8)))  float    v8f;

#define BATCH 4
#define SEQ   1024
#define CDIM  1024
#define HEADS 16
#define HDIM  64
#define ATT_SCALE 0.125f   // 64^-0.5

static __device__ inline v8f wmma_f16(v16h a, v16h b, v8f c) {
  // D = A(16x32 f16) * B(32x16 f16) + C(16x16 f32)
  return __builtin_amdgcn_wmma_f32_16x16x32_f16(false, a, false, b, (short)0, c, false, false);
}

// --- WMMA fragment loaders (contiguous k order k = 16*(lane/16) + s, wave32) ---------
// Both loaders read two contiguous, 16B-aligned 8-half chunks per lane ->
// compiler lowers each to a pair of ds_load_b128.

// A fragment 16x32 (MxK) from row-major f16 buffer with leading dim lda.
static __device__ inline v16h load_a_frag(const _Float16* src, int lda, int m_base, int k_base, int lane) {
  const int m   = m_base + (lane & 15);
  const int sel = (lane >> 4) & 1;
  v16h a;
#pragma unroll
  for (int h = 0; h < 16; ++h) {
    const int k = sel * 16 + h;
    a[h] = src[m * lda + k_base + k];
  }
  return a;
}

// B fragment 32x16 (KxN) where memory holds B transposed: row-major [N][ld].
// (element (k,n) = src[n][k]); per-lane reads are contiguous in k.
static __device__ inline v16h load_b_fragT(const _Float16* src, int ld, int n_base, int k_base, int lane) {
  const int n   = n_base + (lane & 15);
  const int sel = (lane >> 4) & 1;
  v16h b;
#pragma unroll
  for (int h = 0; h < 16; ++h) {
    const int k = k_base + sel * 16 + h;
    b[h] = src[n * ld + k];
  }
  return b;
}

// ---------------------------------------------------------------------------
// GEMM tiling: block 256 threads = 8 waves; tile 128(M) x 64(N) x 32(K).
// Waves in 4x2 grid, each computing a 32x32 sub-tile = 2x2 WMMA fragments.
// A staged [M][K] row-major; B staged [N][K] row-major (fragment-native).
// ---------------------------------------------------------------------------

#define LDA_P 40   // 32 + 8 f16 padding (80 B, multiple of 16)
#define LDB_P 40   // B^T tile leading dim (k + pad)
#define LDQ_P 72   // 64 + 8 f16 padding (144 B, multiple of 16)

#define GEMM_PROLOGUE()                                                        \
  const int tid = threadIdx.x, lane = tid & 31, wave = tid >> 5;               \
  const int wm = wave >> 1, wn = wave & 1;                                     \
  const int m0 = blockIdx.y * 128, n0 = blockIdx.x * 64;                       \
  const int sel = (lane >> 4) & 1, nl = lane & 15;                             \
  v8f cc[2][2];                                                                \
  cc[0][0] = {}; cc[0][1] = {}; cc[1][0] = {}; cc[1][1] = {};

#define GEMM_MMA()                                                             \
  __syncthreads();                                                             \
  {                                                                            \
    v16h a0 = load_a_frag(As, LDA_P, wm * 32, 0, lane);                        \
    v16h a1 = load_a_frag(As, LDA_P, wm * 32 + 16, 0, lane);                   \
    v16h b0 = load_b_fragT(Bs, LDB_P, wn * 32, 0, lane);                       \
    v16h b1 = load_b_fragT(Bs, LDB_P, wn * 32 + 16, 0, lane);                  \
    cc[0][0] = wmma_f16(a0, b0, cc[0][0]);                                     \
    cc[0][1] = wmma_f16(a0, b1, cc[0][1]);                                     \
    cc[1][0] = wmma_f16(a1, b0, cc[1][0]);                                     \
    cc[1][1] = wmma_f16(a1, b1, cc[1][1]);                                     \
  }                                                                            \
  __syncthreads();

// Stage A tile (128x32) from f32 source: 1024 float4 chunks, 4 per thread.
static __device__ inline void stage_a_f32(_Float16* As, const float* X, int m0, int k0, int tid) {
#pragma unroll
  for (int i = 0; i < 4; ++i) {
    int idx = tid + i * 256, r = idx >> 3, c = (idx & 7) * 4;
    float4 f = *(const float4*)(X + (size_t)(m0 + r) * CDIM + k0 + c);
    As[r * LDA_P + c + 0] = (_Float16)f.x;
    As[r * LDA_P + c + 1] = (_Float16)f.y;
    As[r * LDA_P + c + 2] = (_Float16)f.z;
    As[r * LDA_P + c + 3] = (_Float16)f.w;
  }
}

// Stage B^T tile (64 n-rows x 32 k) from f32 weight W[n][k]: 512 chunks, 2/thread.
static __device__ inline void stage_b_f32(_Float16* Bs, const float* W, int n0, int k0, int tid) {
#pragma unroll
  for (int i = 0; i < 2; ++i) {
    int idx = tid + i * 256, n = idx >> 3, c = (idx & 7) * 4;
    float4 f = *(const float4*)(W + (size_t)(n0 + n) * CDIM + k0 + c);
    Bs[n * LDB_P + c + 0] = (_Float16)f.x;
    Bs[n * LDB_P + c + 1] = (_Float16)f.y;
    Bs[n * LDB_P + c + 2] = (_Float16)f.z;
    Bs[n * LDB_P + c + 3] = (_Float16)f.w;
  }
}

// q = xq @ Wq^T  -> Qf in [B][H][N][D] f16
__global__ __launch_bounds__(256) void k_proj_q(const float* __restrict__ X,
                                                const float* __restrict__ W,
                                                _Float16* __restrict__ Qf) {
  __shared__ _Float16 As[128 * LDA_P];
  __shared__ _Float16 Bs[64 * LDB_P];
  GEMM_PROLOGUE();
  for (int k0 = 0; k0 < CDIM; k0 += 32) {
    stage_a_f32(As, X, m0, k0, tid);
    stage_b_f32(Bs, W, n0, k0, tid);
    if (k0 + 32 < CDIM) {
      __builtin_prefetch(X + (size_t)(m0 + (tid >> 1)) * CDIM + k0 + 32 + (tid & 1) * 16, 0, 1);
      if (tid < 64) __builtin_prefetch(W + (size_t)(n0 + tid) * CDIM + k0 + 32, 0, 1);
    }
    GEMM_MMA();
  }
#pragma unroll
  for (int tm = 0; tm < 2; ++tm)
#pragma unroll
    for (int tn = 0; tn < 2; ++tn)
#pragma unroll
      for (int v = 0; v < 8; ++v) {
        int gm = m0 + wm * 32 + tm * 16 + v + 8 * sel;   // = b*SEQ + nq
        int gn = n0 + wn * 32 + tn * 16 + nl;            // = h*HDIM + d
        int b = gm >> 10, nq = gm & 1023;
        int h = gn >> 6, d = gn & 63;
        Qf[(((size_t)(b * HEADS + h)) * SEQ + nq) * HDIM + d] = (_Float16)cc[tm][tn][v];
      }
}

// kv = xk @ Wkv^T -> Kf, Vf in [B][H][N][D] f16
__global__ __launch_bounds__(256) void k_proj_kv(const float* __restrict__ X,
                                                 const float* __restrict__ W,
                                                 _Float16* __restrict__ Kf,
                                                 _Float16* __restrict__ Vf) {
  __shared__ _Float16 As[128 * LDA_P];
  __shared__ _Float16 Bs[64 * LDB_P];
  GEMM_PROLOGUE();
  for (int k0 = 0; k0 < CDIM; k0 += 32) {
    stage_a_f32(As, X, m0, k0, tid);
    stage_b_f32(Bs, W, n0, k0, tid);
    if (k0 + 32 < CDIM) {
      __builtin_prefetch(X + (size_t)(m0 + (tid >> 1)) * CDIM + k0 + 32 + (tid & 1) * 16, 0, 1);
      if (tid < 64) __builtin_prefetch(W + (size_t)(n0 + tid) * CDIM + k0 + 32, 0, 1);
    }
    GEMM_MMA();
  }
#pragma unroll
  for (int tm = 0; tm < 2; ++tm)
#pragma unroll
    for (int tn = 0; tn < 2; ++tn)
#pragma unroll
      for (int v = 0; v < 8; ++v) {
        int gm = m0 + wm * 32 + tm * 16 + v + 8 * sel;
        int gn = n0 + wn * 32 + tn * 16 + nl;            // [0, 2048)
        int b = gm >> 10, nq = gm & 1023;
        float val = cc[tm][tn][v];
        if (gn < CDIM) {
          int h = gn >> 6, d = gn & 63;
          Kf[(((size_t)(b * HEADS + h)) * SEQ + nq) * HDIM + d] = (_Float16)val;
        } else {
          int gn2 = gn - CDIM;
          int h = gn2 >> 6, d = gn2 & 63;
          Vf[(((size_t)(b * HEADS + h)) * SEQ + nq) * HDIM + d] = (_Float16)val;
        }
      }
}

// Flash attention: WG = 64 q-rows of one (b,h); 4 waves, each 16 q-rows.
// Online softmax over key blocks of 32. Output -> Xs f16 [B][N][C].
__global__ __launch_bounds__(128) void k_attn(const _Float16* __restrict__ Qf,
                                              const _Float16* __restrict__ Kf,
                                              const _Float16* __restrict__ Vf,
                                              _Float16* __restrict__ Xs) {
  const int q0 = blockIdx.x * 64;
  const int h  = blockIdx.y;
  const int bb = blockIdx.z;
  const size_t bh = (size_t)(bb * HEADS + h);
  const int tid = threadIdx.x, lane = tid & 31, wave = tid >> 5;
  const int sel = (lane >> 4) & 1, nl = lane & 15;

  __shared__ _Float16 Qs[64 * LDQ_P];         // 64x64 Q tile, [q][d]
  __shared__ _Float16 Ks[32 * LDQ_P];         // 32x64 K block, [kj][d] (B^T for S)
  __shared__ _Float16 VsT[64 * LDB_P];        // V block transposed, [d][kj] (B^T for O)
  __shared__ float    St[4][16 * 33];         // per-wave raw scores 16x32
  __shared__ _Float16 Pt[4][16 * 40];         // per-wave probs 16x32 (A-frag source)
  __shared__ float    Arow[4][16];
  __shared__ float    Lr[4][16];

  // stage Q (4096 halves = 512 x v8h, 4 per thread)
#pragma unroll
  for (int i = 0; i < 4; ++i) {
    int idx = tid + i * 128, r = idx >> 3, c = (idx & 7) * 8;
    v8h qv = *(const v8h*)(Qf + (bh * SEQ + q0 + r) * HDIM + c);
#pragma unroll
    for (int j = 0; j < 8; ++j) Qs[r * LDQ_P + c + j] = qv[j];
  }
  __syncthreads();
  v16h aq0 = load_a_frag(Qs, LDQ_P, wave * 16, 0, lane);
  v16h aq1 = load_a_frag(Qs, LDQ_P, wave * 16, 32, lane);

  v8f o0 = {}, o1 = {}, o2 = {}, o3 = {};
  float mrow = -INFINITY, lrow = 0.f;

  for (int kb = 0; kb < SEQ; kb += 32) {
    __syncthreads();
    // stage K [kj][d] and V^T [d][kj]: 256 v8h chunks each, 2 per thread
#pragma unroll
    for (int i = 0; i < 2; ++i) {
      int idx = tid + i * 128, r = idx >> 3, c = (idx & 7) * 8;
      v8h kv = *(const v8h*)(Kf + (bh * SEQ + kb + r) * HDIM + c);
      v8h vv = *(const v8h*)(Vf + (bh * SEQ + kb + r) * HDIM + c);
#pragma unroll
      for (int j = 0; j < 8; ++j) {
        Ks[r * LDQ_P + c + j]  = kv[j];
        VsT[(c + j) * LDB_P + r] = vv[j];
      }
    }
    if (kb + 32 < SEQ) {
      __builtin_prefetch(Kf + (bh * SEQ + kb + 32 + (tid >> 2)) * HDIM + (tid & 3) * 16, 0, 1);
      __builtin_prefetch(Vf + (bh * SEQ + kb + 32 + (tid >> 2)) * HDIM + (tid & 3) * 16, 0, 1);
    }
    __syncthreads();

    // S(16x32) = Q(16x64) * K^T(64x32); B^T source is Ks[kj][d]
    v8f s0 = {}, s1 = {};
    s0 = wmma_f16(aq0, load_b_fragT(Ks, LDQ_P, 0, 0, lane), s0);
    s1 = wmma_f16(aq0, load_b_fragT(Ks, LDQ_P, 16, 0, lane), s1);
    s0 = wmma_f16(aq1, load_b_fragT(Ks, LDQ_P, 0, 32, lane), s0);
    s1 = wmma_f16(aq1, load_b_fragT(Ks, LDQ_P, 16, 32, lane), s1);

#pragma unroll
    for (int v = 0; v < 8; ++v) {
      int m = v + 8 * sel;
      St[wave][m * 33 + nl]      = s0[v];
      St[wave][m * 33 + 16 + nl] = s1[v];
    }
    __syncthreads();

    if (lane < 16) {
      float rmax = -INFINITY;
#pragma unroll
      for (int c = 0; c < 32; ++c) rmax = fmaxf(rmax, St[wave][lane * 33 + c]);
      rmax *= ATT_SCALE;
      float mnew = fmaxf(mrow, rmax);
      float alpha = __expf(mrow - mnew);
      float rsum = 0.f;
#pragma unroll
      for (int c = 0; c < 32; ++c) {
        float p = __expf(St[wave][lane * 33 + c] * ATT_SCALE - mnew);
        Pt[wave][lane * 40 + c] = (_Float16)p;
        rsum += p;
      }
      lrow = lrow * alpha + rsum;
      mrow = mnew;
      Arow[wave][lane] = alpha;
    }
    __syncthreads();

    float al[8];
#pragma unroll
    for (int v = 0; v < 8; ++v) al[v] = Arow[wave][v + 8 * sel];
#pragma unroll
    for (int v = 0; v < 8; ++v) { o0[v] *= al[v]; o1[v] *= al[v]; o2[v] *= al[v]; o3[v] *= al[v]; }

    // O(16x64) += P(16x32) * V(32x64); B^T source is VsT[d][kj]
    v16h ap = load_a_frag(Pt[wave], 40, 0, 0, lane);
    o0 = wmma_f16(ap, load_b_fragT(VsT, LDB_P, 0, 0, lane), o0);
    o1 = wmma_f16(ap, load_b_fragT(VsT, LDB_P, 16, 0, lane), o1);
    o2 = wmma_f16(ap, load_b_fragT(VsT, LDB_P, 32, 0, lane), o2);
    o3 = wmma_f16(ap, load_b_fragT(VsT, LDB_P, 48, 0, lane), o3);
  }

  if (lane < 16) Lr[wave][lane] = lrow;
  __syncthreads();
#pragma unroll
  for (int v = 0; v < 8; ++v) {
    int m = v + 8 * sel;
    float li = 1.0f / Lr[wave][m];
    int q = q0 + wave * 16 + m;
    size_t rowbase = ((size_t)(bb * SEQ + q)) * CDIM + h * HDIM;
    Xs[rowbase + 0 + nl]  = (_Float16)(o0[v] * li);
    Xs[rowbase + 16 + nl] = (_Float16)(o1[v] * li);
    Xs[rowbase + 32 + nl] = (_Float16)(o2[v] * li);
    Xs[rowbase + 48 + nl] = (_Float16)(o3[v] * li);
  }
}

// attn.sum(heads)/H == (q_full . k_full) * SCALE/H  -> per-batch GEMM, deterministic.
__global__ __launch_bounds__(256) void k_attn_sum(const _Float16* __restrict__ Qf,
                                                  const _Float16* __restrict__ Kf,
                                                  float* __restrict__ AT) {
  __shared__ _Float16 As[128 * LDA_P];
  __shared__ _Float16 Bs[64 * LDB_P];
  const int bb = blockIdx.z;
  GEMM_PROLOGUE();
  for (int k0 = 0; k0 < CDIM; k0 += 32) {
    const int hh = k0 >> 6, dd = k0 & 63;   // 32-wide k-tile sits inside one head's 64-d chunk
    // A: [m][c] from Qf; 512 v8h chunks, 2 per thread
#pragma unroll
    for (int i = 0; i < 2; ++i) {
      int idx = tid + i * 256, r = idx >> 2, c = (idx & 3) * 8;
      v8h qv = *(const v8h*)(Qf + (((size_t)(bb * HEADS + hh)) * SEQ + m0 + r) * HDIM + dd + c);
#pragma unroll
      for (int j = 0; j < 8; ++j) As[r * LDA_P + c + j] = qv[j];
    }
    // B^T: [n][c] from Kf; 256 v8h chunks, 1 per thread
    {
      int idx = tid, n = idx >> 2, c = (idx & 3) * 8;
      v8h kv = *(const v8h*)(Kf + (((size_t)(bb * HEADS + hh)) * SEQ + n0 + n) * HDIM + dd + c);
#pragma unroll
      for (int j = 0; j < 8; ++j) Bs[n * LDB_P + c + j] = kv[j];
    }
    GEMM_MMA();
  }
  const float sc = ATT_SCALE / (float)HEADS;
#pragma unroll
  for (int tm = 0; tm < 2; ++tm)
#pragma unroll
    for (int tn = 0; tn < 2; ++tn)
#pragma unroll
      for (int v = 0; v < 8; ++v) {
        int gm = m0 + wm * 32 + tm * 16 + v + 8 * sel;
        int gn = n0 + wn * 32 + tn * 16 + nl;
        AT[((size_t)bb * SEQ + gm) * SEQ + gn] = cc[tm][tn][v] * sc;
      }
}

// out = x @ Wp^T + bp, stored transposed as [N][B][C]
__global__ __launch_bounds__(256) void k_proj_out(const _Float16* __restrict__ Xh,
                                                  const float* __restrict__ W,
                                                  const float* __restrict__ bias,
                                                  float* __restrict__ Out) {
  __shared__ _Float16 As[128 * LDA_P];
  __shared__ _Float16 Bs[64 * LDB_P];
  GEMM_PROLOGUE();
  for (int k0 = 0; k0 < CDIM; k0 += 32) {
    // A: f16 source; 512 v8h chunks, 2 per thread
#pragma unroll
    for (int i = 0; i < 2; ++i) {
      int idx = tid + i * 256, r = idx >> 2, c = (idx & 3) * 8;
      v8h xv = *(const v8h*)(Xh + (size_t)(m0 + r) * CDIM + k0 + c);
#pragma unroll
      for (int j = 0; j < 8; ++j) As[r * LDA_P + c + j] = xv[j];
    }
    stage_b_f32(Bs, W, n0, k0, tid);
    if (k0 + 32 < CDIM) {
      if (tid < 128) __builtin_prefetch(Xh + (size_t)(m0 + tid) * CDIM + k0 + 32, 0, 1);
      else __builtin_prefetch(W + (size_t)(n0 + (tid & 63)) * CDIM + k0 + 32, 0, 1);
    }
    GEMM_MMA();
  }
#pragma unroll
  for (int tm = 0; tm < 2; ++tm)
#pragma unroll
    for (int tn = 0; tn < 2; ++tn)
#pragma unroll
      for (int v = 0; v < 8; ++v) {
        int gm = m0 + wm * 32 + tm * 16 + v + 8 * sel;   // = b*SEQ + nq
        int gn = n0 + wn * 32 + tn * 16 + nl;
        int b = gm >> 10, nq = gm & 1023;
        Out[((size_t)nq * BATCH + b) * CDIM + gn] = cc[tm][tn][v] + bias[gn];
      }
}

extern "C" void kernel_launch(void* const* d_in, const int* in_sizes, int n_in,
                              void* d_out, int out_size, void* d_ws, size_t ws_size,
                              hipStream_t stream) {
  const float* xq  = (const float*)d_in[0];
  const float* xk  = (const float*)d_in[1];
  // d_in[2] (xv) is unused by the reference (shared KV projection from xk)
  const float* Wq  = (const float*)d_in[3];
  const float* Wkv = (const float*)d_in[4];
  const float* Wp  = (const float*)d_in[5];
  const float* bp  = (const float*)d_in[6];

  float* out_x    = (float*)d_out;                          // [N][B][C]
  float* out_attn = out_x + (size_t)SEQ * BATCH * CDIM;     // [B][N][N]

  const size_t per = (size_t)BATCH * SEQ * CDIM;            // 4M f16 elems each
  _Float16* Qf = (_Float16*)d_ws;
  _Float16* Kf = Qf + per;
  _Float16* Vf = Kf + per;
  _Float16* Xs = Vf + per;                                  // total 32 MB scratch

  k_proj_q  <<<dim3(CDIM / 64, (BATCH * SEQ) / 128), 256, 0, stream>>>(xq, Wq, Qf);
  k_proj_kv <<<dim3((2 * CDIM) / 64, (BATCH * SEQ) / 128), 256, 0, stream>>>(xk, Wkv, Kf, Vf);
  k_attn    <<<dim3(SEQ / 64, HEADS, BATCH), 128, 0, stream>>>(Qf, Kf, Vf, Xs);
  k_attn_sum<<<dim3(SEQ / 64, SEQ / 128, BATCH), 256, 0, stream>>>(Qf, Kf, out_attn);
  k_proj_out<<<dim3(CDIM / 64, (BATCH * SEQ) / 128), 256, 0, stream>>>(Xs, Wp, bp, out_x);
}
